// FMPlus_523986010576
// MI455X (gfx1250) — matrix operand order repaired and checked by reading, not compiled, not run on hardware
//
#include <hip/hip_runtime.h>
#include <math.h>

// ---------------- problem constants ----------------
#define BATCH 4096
#define NF    30
#define ND    32      // embedding dim = WMMA K
#define NPAIR 435     // 30*29/2
#define NPAD  448     // 28 tiles * 16
#define NTILE 28
#define WAVES_PER_BLOCK 4
#define HSTRIDE 456   // 448 + 8: bank-shifts the two half-partial arrays

typedef __attribute__((ext_vector_type(16))) __bf16 v16bf;
typedef __attribute__((ext_vector_type(8)))  float  v8f;

// ---------------- compile-time pair index table (triu_indices(30, k=1)) ----
struct PairTab { unsigned char pi[NPAD]; unsigned char pj[NPAD]; };
static constexpr PairTab makeTab() {
    PairTab t{};
    int p = 0;
    for (int a = 0; a < NF; ++a)
        for (int b = a + 1; b < NF; ++b) { t.pi[p] = (unsigned char)a; t.pj[p] = (unsigned char)b; ++p; }
    for (; p < NPAD; ++p) { t.pi[p] = 0; t.pj[p] = 1; }   // harmless padding rows
    return t;
}
__constant__ PairTab g_tab = makeTab();

// LDS per wave: 960 f32 emb + 2x456 logit partials + 2x456 rowsum partials
#define WAVE_LDS_F (NF*ND + 2*HSTRIDE + 2*HSTRIDE)   // 2784 floats = 11136 B

__global__ __launch_bounds__(WAVES_PER_BLOCK * 32)
void fmplus_attn_kernel(const float* __restrict__ emb,   // [B, 30, 32]
                        const float* __restrict__ w1,    // [32, 32]
                        const float* __restrict__ b1,    // [32]
                        const float* __restrict__ w2,    // [32]
                        float* __restrict__ out)         // [B]
{
    __shared__ float smem[WAVES_PER_BLOCK * WAVE_LDS_F];

    const int lane  = threadIdx.x & 31;
    const int wave  = threadIdx.x >> 5;
    const int batch = blockIdx.x * WAVES_PER_BLOCK + wave;

    float* embLds    = smem + wave * WAVE_LDS_F;         // [30][32] f32
    float* logitsLds = embLds + NF * ND;                 // [2][456] half-partials
    float* sLds      = logitsLds + 2 * HSTRIDE;          // [2][456] half-partials

    const int ln = lane & 15;       // B-matrix column = pair slot within tile
    const int hf = lane >> 4;       // 0: K 0..15 / a 0..7 | 1: K 16..31 / a 8..15

    // ---- stage this batch's embedding into LDS (240 float4, coalesced) ----
    {
        const float4* src = (const float4*)(emb + (size_t)batch * (NF * ND));
        float4* dst = (float4*)embLds;
        #pragma unroll
        for (int t = 0; t < 8; ++t) {
            int idx = lane + t * 32;
            if (idx < 240) dst[idx] = src[idx];
        }
    }

    // ---- A fragments = w1 (16x32 per a-tile), bf16, built once per wave ----
    // 16-bit A layout: lane (ln,hf): elem e -> K=16*hf+e
    v16bf afrag0, afrag1;
    {
        const float* r0 = w1 + ln * ND;          // a = ln        (a-tile 0)
        const float* r1 = w1 + (16 + ln) * ND;   // a = 16 + ln   (a-tile 1)
        const int c0o = 16 * hf, c1o = 16 * hf + 8;
        #pragma unroll
        for (int e = 0; e < 8; ++e) {
            afrag0[e]     = (__bf16)r0[c0o + e];
            afrag0[8 + e] = (__bf16)r0[c1o + e];
            afrag1[e]     = (__bf16)r1[c0o + e];
            afrag1[8 + e] = (__bf16)r1[c1o + e];
        }
    }

    // ---- per-lane epilogue coefficients: C elem r maps to a = r + 8*hf (+16) ----
    alignas(16) float bb0[8], bb1[8], ww0[8], ww1[8];
    *(float4*)(bb0)     = *(const float4*)(b1 + 8 * hf);
    *(float4*)(bb0 + 4) = *(const float4*)(b1 + 8 * hf + 4);
    *(float4*)(bb1)     = *(const float4*)(b1 + 16 + 8 * hf);
    *(float4*)(bb1 + 4) = *(const float4*)(b1 + 16 + 8 * hf + 4);
    *(float4*)(ww0)     = *(const float4*)(w2 + 8 * hf);
    *(float4*)(ww0 + 4) = *(const float4*)(w2 + 8 * hf + 4);
    *(float4*)(ww1)     = *(const float4*)(w2 + 16 + 8 * hf);
    *(float4*)(ww1 + 4) = *(const float4*)(w2 + 16 + 8 * hf + 4);

    __syncthreads();   // emb staging visible

    // ---- 28 tiles of 16 pairs: B-frag = pair^T in regs -> 2 x WMMA ----
    for (int tile = 0; tile < NTILE; ++tile) {
        const int p  = tile * 16 + ln;
        const int pi = g_tab.pi[p];
        const int pj = g_tab.pj[p];

        // 16-bit B layout (32x16): lane (ln,hf) elem e = pair[p=ln][d = 16*hf + e]
        const float* ri = embLds + pi * ND + 16 * hf;
        const float* rj = embLds + pj * ND + 16 * hf;

        alignas(16) float fi[16], fj[16];
        *(float4*)(fi + 0)  = *(const float4*)(ri + 0);
        *(float4*)(fi + 4)  = *(const float4*)(ri + 4);
        *(float4*)(fi + 8)  = *(const float4*)(ri + 8);
        *(float4*)(fi + 12) = *(const float4*)(ri + 12);
        *(float4*)(fj + 0)  = *(const float4*)(rj + 0);
        *(float4*)(fj + 4)  = *(const float4*)(rj + 4);
        *(float4*)(fj + 8)  = *(const float4*)(rj + 8);
        *(float4*)(fj + 12) = *(const float4*)(rj + 12);

        v16bf bfrag;
        float sp = 0.f;                 // K-half partial of rowsum s[p]
        #pragma unroll
        for (int k = 0; k < 16; ++k) {
            float pr = fi[k] * fj[k];
            sp += pr;
            bfrag[k] = (__bf16)pr;
        }

        // h^T tiles: C[M = a, N = pair];  K = 32 = D in a single bf16 WMMA
        v8f c0 = {}, c1 = {};
        c0 = __builtin_amdgcn_wmma_f32_16x16x32_bf16(false, afrag0, false, bfrag,
                                                     (short)0, c0, false, false);
        c1 = __builtin_amdgcn_wmma_f32_16x16x32_bf16(false, afrag1, false, bfrag,
                                                     (short)0, c1, false, false);

        // logit half-partial: sum over the 16 a-units this lane holds (registers!)
        float lp = 0.f;
        #pragma unroll
        for (int r = 0; r < 8; ++r) {
            lp += fmaxf(c0[r] + bb0[r], 0.f) * ww0[r];
            lp += fmaxf(c1[r] + bb1[r], 0.f) * ww1[r];
        }

        // non-divergent stores: both halves store partials, combined at read time
        sLds[hf * HSTRIDE + p]      = sp;
        logitsLds[hf * HSTRIDE + p] = lp;
    }

    __syncthreads();   // partials visible (also orders wave-local LDS)

    // ---- softmax-weighted reduction: out = (sum exp(l-m)*s) / (sum exp(l-m)) ----
    float lv[14], sv[14];
    #pragma unroll
    for (int t = 0; t < 14; ++t) {
        int p = lane + t * 32;
        if (p < NPAIR) {
            lv[t] = logitsLds[p] + logitsLds[HSTRIDE + p];
            sv[t] = sLds[p] + sLds[HSTRIDE + p];
        } else {
            lv[t] = -INFINITY;
            sv[t] = 0.f;
        }
    }

    float m = -INFINITY;
    #pragma unroll
    for (int t = 0; t < 14; ++t) m = fmaxf(m, lv[t]);
    #pragma unroll
    for (int off = 16; off >= 1; off >>= 1) m = fmaxf(m, __shfl_xor(m, off));

    float se = 0.f, acc = 0.f;
    #pragma unroll
    for (int t = 0; t < 14; ++t) {
        float e = __expf(lv[t] - m);
        se  += e;
        acc += e * sv[t];
    }
    #pragma unroll
    for (int off = 16; off >= 1; off >>= 1) {
        se  += __shfl_xor(se,  off);
        acc += __shfl_xor(acc, off);
    }

    if (lane == 0) out[batch] = acc / se;
}

extern "C" void kernel_launch(void* const* d_in, const int* in_sizes, int n_in,
                              void* d_out, int out_size, void* d_ws, size_t ws_size,
                              hipStream_t stream) {
    const float* emb = (const float*)d_in[0];   // [4096,30,32]
    const float* w1  = (const float*)d_in[1];   // [32,32]
    const float* b1  = (const float*)d_in[2];   // [32]
    const float* w2  = (const float*)d_in[3];   // [32]
    float* out = (float*)d_out;                 // [4096,1]

    dim3 grid(BATCH / WAVES_PER_BLOCK);         // 1024 blocks
    dim3 block(WAVES_PER_BLOCK * 32);           // 128 threads = 4 waves
    fmplus_attn_kernel<<<grid, block, 0, stream>>>(emb, w1, b1, w2, out);
}
